// MoE_84164179132797
// MI455X (gfx1250) — matrix operand order repaired and checked by reading, not compiled, not run on hardware
//
#include <hip/hip_runtime.h>

// ---------------------------------------------------------------------------
// MoE forward for MI455X (gfx1250, wave32, WMMA bf16 16x16x32, f32 accum)
// B=65536, IN_DIM=512, HID=256, EXP_HID=100 (padded to 112), OUT_DIM=1, N_EXP=16
// Shared GEMM uses double-buffered LDS with GLOBAL_LOAD_ASYNC_TO_LDS_B128
// (ASYNCcnt) for the bf16 B tile, register-converted A tile, and batched
// 8x v_wmma_f32_16x16x32_bf16 per K-step.
// ---------------------------------------------------------------------------

#define B_N      65536
#define IN_DIM   512
#define HID      256
#define EXP_HID  100
#define EXP_PAD  112        // 7 tiles of 16
#define N_EXP    16

typedef __attribute__((ext_vector_type(16))) __bf16 v16bf;
typedef __attribute__((ext_vector_type(8)))  __bf16 v8bf;
typedef __attribute__((ext_vector_type(8)))  float  v8f;
typedef __attribute__((ext_vector_type(4)))  float  v4f;

static __device__ __forceinline__ v16bf cat8(v8bf lo, v8bf hi) {
    return __builtin_shufflevector(lo, hi, 0,1,2,3,4,5,6,7,8,9,10,11,12,13,14,15);
}

// LDS byte offset of a __shared__ pointer (low 32 bits of the generic address)
template <typename T>
static __device__ __forceinline__ unsigned lds_off(T* p) {
    return (unsigned)(unsigned long long)(void*)p;
}

// async global->LDS copy of 16 bytes (ASYNCcnt tracked)
static __device__ __forceinline__ void async_b128(unsigned ldsDst, const void* gsrc) {
    asm volatile("global_load_async_to_lds_b128 %0, %1, off"
                 :: "v"(ldsDst), "v"(gsrc) : "memory");
}
static __device__ __forceinline__ void wait_async0() {
    asm volatile("s_wait_asynccnt 0x0" ::: "memory");
}

// ---------------------------------------------------------------------------
// Workspace layout (bytes)
// ---------------------------------------------------------------------------
#define WS_HBUF_OFF   ((size_t)0)                               // [B][HID] bf16   = 32 MiB
#define WS_WST_OFF    (WS_HBUF_OFF + (size_t)B_N*HID*2)         // [HID][IN] bf16  = 256 KiB
#define WS_W1T_OFF    (WS_WST_OFF  + (size_t)HID*IN_DIM*2)      // [E][112][HID] bf16
#define WS_PERM_OFF   (WS_W1T_OFF  + (size_t)N_EXP*EXP_PAD*HID*2) // [B] int
#define WS_CNT_OFF    (WS_PERM_OFF + (size_t)B_N*4)             // 16 int
#define WS_OFFS_OFF   (WS_CNT_OFF  + 64)                        // 16 int
#define WS_CURS_OFF   (WS_OFFS_OFF + 64)                        // 16 int

// ---------------------------------------------------------------------------
// Prep kernels
// ---------------------------------------------------------------------------
__global__ void zero_meta_kernel(int* cnt, int* curs) {
    int i = threadIdx.x;
    if (i < N_EXP) { cnt[i] = 0; curs[i] = 0; }
}

// Wst[n][k] = bf16(Ws[k][n])
__global__ void convert_ws_kernel(const float* __restrict__ Ws, __bf16* __restrict__ Wst) {
    int i = blockIdx.x * blockDim.x + threadIdx.x;
    if (i >= HID * IN_DIM) return;
    int n = i / IN_DIM, k = i % IN_DIM;
    Wst[i] = (__bf16)Ws[(size_t)k * HID + n];
}

// W1t[e][n][k] = bf16(W1[e][k][n]), zero-padded for n in [100,112)
__global__ void convert_w1_kernel(const float* __restrict__ W1, __bf16* __restrict__ W1t) {
    int i = blockIdx.x * blockDim.x + threadIdx.x;
    if (i >= N_EXP * EXP_PAD * HID) return;
    int e = i / (EXP_PAD * HID);
    int r = i % (EXP_PAD * HID);
    int n = r / HID, k = r % HID;
    float v = (n < EXP_HID) ? W1[((size_t)e * HID + k) * EXP_HID + n] : 0.0f;
    W1t[i] = (__bf16)v;
}

// ---------------------------------------------------------------------------
// Expert bucketing: count -> exclusive scan -> scatter
// ---------------------------------------------------------------------------
__global__ void count_kernel(const int* __restrict__ idx, int* __restrict__ cnt) {
    int i = blockIdx.x * blockDim.x + threadIdx.x;
    if (i < B_N) atomicAdd(&cnt[idx[i] & (N_EXP - 1)], 1);
}

__global__ void scan_kernel(const int* __restrict__ cnt, int* __restrict__ offs) {
    if (threadIdx.x == 0) {
        int a = 0;
        for (int e = 0; e < N_EXP; ++e) { offs[e] = a; a += cnt[e]; }
    }
}

__global__ void scatter_kernel(const int* __restrict__ idx, const int* __restrict__ offs,
                               int* __restrict__ curs, int* __restrict__ perm) {
    int i = blockIdx.x * blockDim.x + threadIdx.x;
    if (i >= B_N) return;
    int e = idx[i] & (N_EXP - 1);
    int p = atomicAdd(&curs[e], 1);
    perm[offs[e] + p] = i;
}

// ---------------------------------------------------------------------------
// Shared layer GEMM:  h = relu(x @ Ws + bs)  ->  bf16 hbuf[B][HID]
// Block: 256 threads (8 waves). Block tile: M=64, N=256 (full), K-step 32.
// Double-buffered LDS; B tile staged by async-to-LDS, A tile via registers.
// ---------------------------------------------------------------------------
#define LDA 40   // padded row stride (elements) for LDS tiles (80B, 16B aligned)
#define KSTEPS (IN_DIM / 32)

__global__ __launch_bounds__(256, 2)
void shared_gemm_kernel(const float* __restrict__ x, const __bf16* __restrict__ Wst,
                        const float* __restrict__ bs, __bf16* __restrict__ hbuf) {
    __shared__ __bf16 As[2][64 * LDA];    // [m][k]
    __shared__ __bf16 Bs[2][256 * LDA];   // [n][k]

    const int tid   = threadIdx.x;
    const int wave  = tid >> 5;
    const int lane  = tid & 31;
    const int half  = lane >> 4;
    const int l16   = lane & 15;
    const int mtile = wave & 3;
    const int nbase = (wave >> 2) * 128;
    const int m0    = blockIdx.x * 64;

    // staging coordinates
    const int am = tid >> 2;            // 0..63  : A row
    const int ak = (tid & 3) * 8;       // 0,8,16,24 : A k-chunk
    const float*  asrc = x + (size_t)(m0 + am) * IN_DIM + ak;
    const __bf16* bsrc = Wst + (size_t)tid * IN_DIM;

    __bf16* aDst[2] = { &As[0][am * LDA + ak], &As[1][am * LDA + ak] };
    unsigned bDst[2] = { lds_off(&Bs[0][tid * LDA]), lds_off(&Bs[1][tid * LDA]) };

    // ---- issue async B-tile copy for K-step k0 into buffer `buf` ----
    auto issueB = [&](int buf, int k0) {
#pragma unroll
        for (int r = 0; r < 4; ++r)
            async_b128(bDst[buf] + r * 16, bsrc + k0 + r * 8);
    };
    // ---- load + convert A chunk (8 f32 -> 8 bf16) for K-step k0 ----
    auto loadA = [&](int k0) -> v8bf {
        v4f f0 = *(const v4f*)(asrc + k0);
        v4f f1 = *(const v4f*)(asrc + k0 + 4);
        v8bf h;
#pragma unroll
        for (int j = 0; j < 4; ++j) { h[j] = (__bf16)f0[j]; h[4 + j] = (__bf16)f1[j]; }
        return h;
    };

    v8f acc[8];
#pragma unroll
    for (int i = 0; i < 8; ++i) acc[i] = (v8f)0.0f;

    // prologue: stage buffer 0
    issueB(0, 0);
    *(v8bf*)aDst[0] = loadA(0);
    wait_async0();
    __syncthreads();

    for (int kk = 0; kk < KSTEPS; ++kk) {
        const int cur = kk & 1, nxt = cur ^ 1;
        const int k0  = kk * 32;

        // issue next tile's loads before compute so they overlap the WMMAs
        v8bf aNext;
        if (kk + 1 < KSTEPS) {
            issueB(nxt, k0 + 32);
            aNext = loadA(k0 + 32);
        }

        // A fragment (contiguous k order: k = 16*half + s, matching B)
        const __bf16* arow = &As[cur][(mtile * 16 + l16) * LDA];
        v8bf alo = *(const v8bf*)(arow + 16 * half);
        v8bf ahi = *(const v8bf*)(arow + 16 * half + 8);
        v16bf afrag = cat8(alo, ahi);

        // preload all 8 B fragments, then run 8 WMMAs back-to-back
        v16bf bfrag[8];
#pragma unroll
        for (int nt = 0; nt < 8; ++nt)
            bfrag[nt] = *(const v16bf*)(&Bs[cur][(nbase + nt * 16 + l16) * LDA + 16 * half]);
#pragma unroll
        for (int nt = 0; nt < 8; ++nt)
            acc[nt] = __builtin_amdgcn_wmma_f32_16x16x32_bf16(
                false, afrag, false, bfrag[nt], (short)0, acc[nt], false, false);

        if (kk + 1 < KSTEPS) *(v8bf*)aDst[nxt] = aNext;
        wait_async0();          // next buffer's async B copy complete
        __syncthreads();        // + compiler-inserted dscnt wait for A stores
    }

    // ---- epilogue: bias + ReLU, store bf16 h ----
#pragma unroll
    for (int nt = 0; nt < 8; ++nt) {
        int ng = nbase + nt * 16 + l16;
        float bv = bs[ng];
#pragma unroll
        for (int r = 0; r < 8; ++r) {
            int gm = m0 + mtile * 16 + r + 8 * half;
            float h = acc[nt][r] + bv;
            h = h > 0.0f ? h : 0.0f;
            hbuf[(size_t)gm * HID + ng] = (__bf16)h;
        }
    }
}

// ---------------------------------------------------------------------------
// Expert MLP: per 16-sample tile of one expert (one wave per block).
//   h1 = relu(h @ W1t[e] + b1[e]) ;  out = h1 . W2[e] + b2[e]
// ---------------------------------------------------------------------------
__global__ __launch_bounds__(32)
void expert_kernel(const __bf16* __restrict__ hbuf, const __bf16* __restrict__ W1t,
                   const float* __restrict__ b1, const float* __restrict__ W2,
                   const float* __restrict__ b2, const int* __restrict__ perm,
                   const int* __restrict__ offs, const int* __restrict__ cnt,
                   float* __restrict__ out) {
    const int e = blockIdx.y;
    const int t = blockIdx.x;
    const int c = cnt[e];
    if (t * 16 >= c) return;                 // whole-wave exit: EXEC stays full
    const int start = offs[e];
    const int rows  = min(16, c - t * 16);

    const int lane = threadIdx.x;
    const int half = lane >> 4;
    const int l16  = lane & 15;

    // gather this lane's A row (clamped so all 32 lanes stay active for WMMA)
    const int mrow = min(l16, rows - 1);
    const int samp = perm[start + t * 16 + mrow];
    const __bf16* arow = hbuf + (size_t)samp * HID;

    v16bf afrag[8];
#pragma unroll
    for (int kk = 0; kk < 8; ++kk) {
        int k0 = kk * 32;
        v8bf lo = *(const v8bf*)(arow + k0 + 16 * half);
        v8bf hi = *(const v8bf*)(arow + k0 + 16 * half + 8);
        afrag[kk] = cat8(lo, hi);
    }

    float outAcc[8];
#pragma unroll
    for (int r = 0; r < 8; ++r) outAcc[r] = 0.0f;

#pragma unroll
    for (int nt = 0; nt < 7; ++nt) {
        const __bf16* bbase =
            W1t + ((size_t)e * EXP_PAD + nt * 16 + l16) * HID + 16 * half;

        // preload all 8 B fragments, then 8 back-to-back WMMAs
        v16bf bfrag[8];
#pragma unroll
        for (int kk = 0; kk < 8; ++kk)
            bfrag[kk] = *(const v16bf*)(bbase + kk * 32);

        v8f acc = (v8f)0.0f;
#pragma unroll
        for (int kk = 0; kk < 8; ++kk)
            acc = __builtin_amdgcn_wmma_f32_16x16x32_bf16(
                false, afrag[kk], false, bfrag[kk], (short)0, acc, false, false);

        // fuse bias + ReLU + layer-2 dot (W2 column, OUT_DIM=1)
        int ng = nt * 16 + l16;
        float b1v = (ng < EXP_HID) ? b1[e * EXP_HID + ng] : 0.0f;
        float w2v = (ng < EXP_HID) ? W2[e * EXP_HID + ng] : 0.0f;
#pragma unroll
        for (int r = 0; r < 8; ++r) {
            float h1 = acc[r] + b1v;
            h1 = h1 > 0.0f ? h1 : 0.0f;
            outAcc[r] += h1 * w2v;
        }
    }

    const float b2v = b2[e];
#pragma unroll
    for (int r = 0; r < 8; ++r) {
        float v = outAcc[r];
        v += __shfl_xor(v, 1, 16);
        v += __shfl_xor(v, 2, 16);
        v += __shfl_xor(v, 4, 16);
        v += __shfl_xor(v, 8, 16);
        if (l16 == 0) {
            int m = r + 8 * half;
            if (m < rows) out[perm[start + t * 16 + m]] = v + b2v;
        }
    }
}

// ---------------------------------------------------------------------------
// Launcher
// ---------------------------------------------------------------------------
extern "C" void kernel_launch(void* const* d_in, const int* in_sizes, int n_in,
                              void* d_out, int out_size, void* d_ws, size_t ws_size,
                              hipStream_t stream) {
    const float* x  = (const float*)d_in[0];
    const int*   idx= (const int*)  d_in[1];
    const float* Ws = (const float*)d_in[2];
    const float* bs = (const float*)d_in[3];
    const float* W1 = (const float*)d_in[4];
    const float* b1 = (const float*)d_in[5];
    const float* W2 = (const float*)d_in[6];
    const float* b2 = (const float*)d_in[7];
    float* out = (float*)d_out;

    char* ws = (char*)d_ws;
    __bf16* hbuf = (__bf16*)(ws + WS_HBUF_OFF);
    __bf16* Wst  = (__bf16*)(ws + WS_WST_OFF);
    __bf16* W1t  = (__bf16*)(ws + WS_W1T_OFF);
    int* perm = (int*)(ws + WS_PERM_OFF);
    int* cnt  = (int*)(ws + WS_CNT_OFF);
    int* offs = (int*)(ws + WS_OFFS_OFF);
    int* curs = (int*)(ws + WS_CURS_OFF);

    // prep
    zero_meta_kernel<<<1, 32, 0, stream>>>(cnt, curs);
    convert_ws_kernel<<<(HID * IN_DIM + 255) / 256, 256, 0, stream>>>(Ws, Wst);
    convert_w1_kernel<<<(N_EXP * EXP_PAD * HID + 255) / 256, 256, 0, stream>>>(W1, W1t);

    // bucket by expert
    count_kernel<<<B_N / 256, 256, 0, stream>>>(idx, cnt);
    scan_kernel<<<1, 32, 0, stream>>>(cnt, offs);
    scatter_kernel<<<B_N / 256, 256, 0, stream>>>(idx, offs, curs, perm);

    // shared layer GEMM (bf16 WMMA, f32 accum, async-to-LDS double buffering)
    shared_gemm_kernel<<<B_N / 64, 256, 0, stream>>>(x, Wst, bs, hbuf);

    // expert MLP (bf16 WMMA + fused layer 2)
    dim3 egrid(B_N / 16, N_EXP);
    expert_kernel<<<egrid, 32, 0, stream>>>(hbuf, W1t, b1, W2, b2, perm, offs, cnt, out);
}
